// MoE_15436112462202
// MI455X (gfx1250) — compile-verified, matrix operand order checked by reading, not run on hardware
//
#include <hip/hip_runtime.h>
#include <stdint.h>

// ---------------------------------------------------------------------------
// Soft-MoE forward for MI455X (gfx1250):
//  - all 6 contractions via v_wmma_f32_16x16x32_bf16 (wave32)
//  - double-buffered LDS tiles, b128 staging
//  - async global->LDS copies (ASYNCcnt) for K-contiguous operands
// ---------------------------------------------------------------------------

typedef __attribute__((ext_vector_type(16))) __bf16 v16bf;
typedef __attribute__((ext_vector_type(8)))  __bf16 bf16x8;
typedef __attribute__((ext_vector_type(8)))  float  v8f;
typedef __attribute__((ext_vector_type(4)))  int    v4i;

// Problem constants (from reference)
constexpr int Bb   = 16;
constexpr int Tt   = 2048;
constexpr int Cc   = 1024;
constexpr int Ee   = 16;
constexpr int Hh   = 2048;
constexpr int CAP  = Tt / Ee;     // 128
constexpr int ECAP = Ee * CAP;    // 2048
constexpr int BCAP = Bb * CAP;    // 2048

constexpr int BM = 128;
constexpr int BN = 128;

// Async global->LDS availability (device pass only).
#if defined(__AMDGCN__) && __has_builtin(__builtin_amdgcn_global_load_async_to_lds_b128)
#define ASYNC_AVAIL 1
#else
#define ASYNC_AVAIL 0
#endif

#if ASYNC_AVAIL
__device__ __forceinline__ void async_cp_b128(const void* g, void* l) {
  // param0: v4i* in global AS (prints as "__device__" in HIP clang),
  // param1: v4i* in LDS AS(3), then imm offset + cpol.
  __builtin_amdgcn_global_load_async_to_lds_b128(
      (__attribute__((address_space(1))) v4i*)g,
      (__attribute__((address_space(3))) v4i*)l, 0, 0);
}
__device__ __forceinline__ void wait_async0() {
#if __has_builtin(__builtin_amdgcn_s_wait_asynccnt)
  __builtin_amdgcn_s_wait_asynccnt(0);
#else
  asm volatile("s_wait_asynccnt 0x0" ::: "memory");
#endif
}
#endif

__device__ __forceinline__ __bf16 f2bf(float f) {
  // round-to-nearest-even f32 -> bf16 (bit ops only)
  unsigned u = __builtin_bit_cast(unsigned, f);
  u += 0x7FFFu + ((u >> 16) & 1u);
  unsigned short h = (unsigned short)(u >> 16);
  return __builtin_bit_cast(__bf16, h);
}

// WMMA 16x32 bf16 operand fragment from an LDS tile row (pitch LDP):
//   lanes 0-15 : K 0..7 (elems 0..7),  K 16..23 (elems 8..15)
//   lanes 16-31: K 8..15,              K 24..31
__device__ __forceinline__ v16bf load_frag(const __bf16* rowp, int lane) {
  const int off = (lane >> 4) << 3;             // 0 or 8
  bf16x8 lo = *(const bf16x8*)(rowp + off);
  bf16x8 hi = *(const bf16x8*)(rowp + 16 + off);
  v16bf f;
#pragma unroll
  for (int i = 0; i < 8; ++i) { f[i] = lo[i]; f[i + 8] = hi[i]; }
  return f;
}

enum { M_ROUTER = 0, M_DISPATCH = 1, M_GLU = 2, M_PROJ = 3, M_COMBINE = 4 };

template <int MODE>
__global__ __launch_bounds__(256, 1)
void gemm_bf16(const __bf16* __restrict__ A0,
               const __bf16* __restrict__ Bm0,
               const __bf16* __restrict__ Bm1,
               float*  __restrict__ OutF,
               __bf16* __restrict__ OutH,
               int Kdim) {
  constexpr bool GLU = (MODE == M_GLU);
  constexpr int  BKT = GLU ? 32 : 64;       // K per stage (GLU: 3 operands)
  constexpr int  LDP = BKT + 8;             // LDS pitch (bytes%16==0)
  constexpr int  CH  = BKT / 16;            // b128 chunks per thread/operand
  constexpr int  CPR = BKT / 8;             // chunks per row (K-contig layout)
  constexpr int  KS  = BKT / 32;            // WMMA k-steps per stage

  const int tid  = threadIdx.x;
  const int lane = tid & 31;
  const int wave = tid >> 5;
  const int z    = blockIdx.z;              // batch or expert
  const int row0 = blockIdx.y * BM;
  const int col0 = blockIdx.x * BN;

  __shared__ __bf16 sA[2][BM * LDP];
  __shared__ __bf16 sB[2][BN * LDP];
  __shared__ __bf16 sB2[GLU ? 2 : 1][GLU ? BN * LDP : 8];

  auto aAddr = [&](int r, int k) -> const __bf16* {
    if constexpr (MODE == M_ROUTER)   return A0 + ((size_t)z * Tt   + r) * Cc   + k; // x[b]
    if constexpr (MODE == M_DISPATCH) return A0 + ((size_t)z * Tt   + k) * ECAP + r; // disp^T
    if constexpr (MODE == M_GLU)      return A0 + ((size_t)z * BCAP + r) * Cc   + k; // ei[e]
    if constexpr (MODE == M_PROJ)     return A0 + ((size_t)z * BCAP + r) * Hh   + k; // h[e]
    return A0 + ((size_t)z * Tt + r) * ECAP + k;                                     // collect[b]
  };
  auto bAddr = [&](const __bf16* base, int k, int n) -> const __bf16* {
    if constexpr (MODE == M_ROUTER)   return base + (size_t)n * Cc + k;              // Wr[n][k]
    if constexpr (MODE == M_DISPATCH) return base + ((size_t)z * Tt + k) * Cc + n;   // x[b][k][n]
    if constexpr (MODE == M_GLU)      return base + ((size_t)z * Cc + k) * Hh + n;   // W[e][k][n]
    if constexpr (MODE == M_PROJ)     return base + ((size_t)z * Hh + k) * Cc + n;   // Wp[e][k][n]
    // COMBINE: eo grouped [E][B*CAP][C]; k = e*CAP + s, batch = z
    return base + ((size_t)(k / CAP) * BCAP + (size_t)z * CAP + (k % CAP)) * Cc + n;
  };

  constexpr bool A_KC = (MODE != M_DISPATCH);  // A is K-contiguous in memory
  constexpr bool B_KC = (MODE == M_ROUTER);    // B is K-contiguous in memory

  bf16x8 rA[CH], rB[CH], rB2[GLU ? CH : 1];

  // ---- stage issue/commit ------------------------------------------------
  auto issueA = [&](int k0, __bf16* buf) {
#pragma unroll
    for (int c = 0; c < CH; ++c) {
      const int idx = tid + c * 256;
      if constexpr (A_KC) {
        const int rr = idx / CPR, kv = (idx % CPR) * 8;
        const __bf16* gp = aAddr(row0 + rr, k0 + kv);
#if ASYNC_AVAIL
        async_cp_b128(gp, &buf[rr * LDP + kv]);   // direct global->LDS, ASYNCcnt
#else
        rA[c] = *(const bf16x8*)gp;
#endif
      } else {
        const int kk = idx >> 4, rr = (idx & 15) * 8;
        rA[c] = *(const bf16x8*)aAddr(row0 + rr, k0 + kk);
      }
    }
  };
  auto commitA = [&](__bf16* buf) {
#pragma unroll
    for (int c = 0; c < CH; ++c) {
      const int idx = tid + c * 256;
      if constexpr (A_KC) {
#if !ASYNC_AVAIL
        const int rr = idx / CPR, kv = (idx % CPR) * 8;
        *(bf16x8*)&buf[rr * LDP + kv] = rA[c];
#endif
      } else {
        const int kk = idx >> 4, rr = (idx & 15) * 8;
#pragma unroll
        for (int j = 0; j < 8; ++j) buf[(rr + j) * LDP + kk] = rA[c][j];
      }
    }
  };
  auto issueB = [&](const __bf16* base, int k0, __bf16* buf, bf16x8* regs) {
#pragma unroll
    for (int c = 0; c < CH; ++c) {
      const int idx = tid + c * 256;
      if constexpr (B_KC) {
        const int rr = idx / CPR, kv = (idx % CPR) * 8;
        const __bf16* gp = bAddr(base, k0 + kv, col0 + rr);
#if ASYNC_AVAIL
        async_cp_b128(gp, &buf[rr * LDP + kv]);
#else
        regs[c] = *(const bf16x8*)gp;
#endif
      } else {
        const int kk = idx >> 4, rr = (idx & 15) * 8;
        regs[c] = *(const bf16x8*)bAddr(base, k0 + kk, col0 + rr);
      }
    }
  };
  auto commitB = [&](__bf16* buf, const bf16x8* regs) {
#pragma unroll
    for (int c = 0; c < CH; ++c) {
      const int idx = tid + c * 256;
      if constexpr (B_KC) {
#if !ASYNC_AVAIL
        const int rr = idx / CPR, kv = (idx % CPR) * 8;
        *(bf16x8*)&buf[rr * LDP + kv] = regs[c];
#endif
      } else {
        const int kk = idx >> 4, rr = (idx & 15) * 8;
#pragma unroll
        for (int j = 0; j < 8; ++j) buf[(rr + j) * LDP + kk] = regs[c][j];
      }
    }
  };

  v8f acc[2][4] = {};
  v8f acc2[GLU ? 2 : 1][GLU ? 4 : 1] = {};

  const int wm   = (wave & 3) * 32;   // wave M offset in block tile
  const int wn   = (wave >> 2) * 64;  // wave N offset
  const int mrow = lane & 15;

  // ---- prologue: fill buffer 0 ------------------------------------------
  issueA(0, sA[0]);
  issueB(Bm0, 0, sB[0], rB);
  if constexpr (GLU) issueB(Bm1, 0, sB2[0], rB2);
  commitA(sA[0]);
  commitB(sB[0], rB);
  if constexpr (GLU) commitB(sB2[0], rB2);
#if ASYNC_AVAIL
  wait_async0();
#endif
  __syncthreads();

  const int nk = Kdim / BKT;
  for (int kt = 0; kt < nk; ++kt) {
    const int cur = kt & 1;
    const int nxt = cur ^ 1;

    // issue loads for next stage (async lands directly in the other buffer)
    if (kt + 1 < nk) {
      const int k1 = (kt + 1) * BKT;
      issueA(k1, sA[nxt]);
      issueB(Bm0, k1, sB[nxt], rB);
      if constexpr (GLU) issueB(Bm1, k1, sB2[nxt], rB2);
      if (kt + 2 < nk)  // L2 warm-up for the tile after next
        __builtin_prefetch(aAddr(row0 + (tid >> 1), (kt + 2) * BKT), 0, 1);
    }

    // ---- WMMA compute from current buffer
#pragma unroll
    for (int ks = 0; ks < KS; ++ks) {
      const int ko = ks * 32;
      v16bf af[2], bfr[4];
#pragma unroll
      for (int mi = 0; mi < 2; ++mi)
        af[mi] = load_frag(&sA[cur][(wm + mi * 16 + mrow) * LDP + ko], lane);
#pragma unroll
      for (int ni = 0; ni < 4; ++ni)
        bfr[ni] = load_frag(&sB[cur][(wn + ni * 16 + mrow) * LDP + ko], lane);
#pragma unroll
      for (int mi = 0; mi < 2; ++mi)
#pragma unroll
        for (int ni = 0; ni < 4; ++ni)
          acc[mi][ni] = __builtin_amdgcn_wmma_f32_16x16x32_bf16(
              false, af[mi], false, bfr[ni], (short)0, acc[mi][ni], false, false);
      if constexpr (GLU) {
        v16bf bfr2[4];
#pragma unroll
        for (int ni = 0; ni < 4; ++ni)
          bfr2[ni] = load_frag(&sB2[cur][(wn + ni * 16 + mrow) * LDP + ko], lane);
#pragma unroll
        for (int mi = 0; mi < 2; ++mi)
#pragma unroll
          for (int ni = 0; ni < 4; ++ni)
            acc2[mi][ni] = __builtin_amdgcn_wmma_f32_16x16x32_bf16(
                false, af[mi], false, bfr2[ni], (short)0, acc2[mi][ni], false, false);
      }
    }

    // commit staged regs into next buffer, drain async, swap
    if (kt + 1 < nk) {
      commitA(sA[nxt]);
      commitB(sB[nxt], rB);
      if constexpr (GLU) commitB(sB2[nxt], rB2);
#if ASYNC_AVAIL
      wait_async0();
#endif
      __syncthreads();
    }
  }

  // ---- epilogue. C frag: VGPR r, lanes 0-15 -> M=r, N=lane;
  //      lanes 16-31 -> M=8+r, N=lane-16.
  const int halfm = (lane >> 4) << 3;
  const int ncol  = lane & 15;
#pragma unroll
  for (int mi = 0; mi < 2; ++mi)
#pragma unroll
    for (int ni = 0; ni < 4; ++ni)
#pragma unroll
      for (int r = 0; r < 8; ++r) {
        const int row = row0 + wm + mi * 16 + halfm + r;
        const int col = col0 + wn + ni * 16 + ncol;
        const float v = acc[mi][ni][r];
        if constexpr (MODE == M_ROUTER) {
          OutF[((size_t)z * Tt + row) * ECAP + col] = v;
        } else if constexpr (MODE == M_DISPATCH) {
          OutH[((size_t)(row / CAP) * BCAP + (size_t)z * CAP + (row % CAP)) * Cc + col] = f2bf(v);
        } else if constexpr (MODE == M_GLU) {
          const float g = v;                  // ei @ w_gate
          const float f = acc2[mi][ni][r];    // ei @ w_fc
          const float h = (g / (1.0f + __expf(-g))) * f;   // silu(g)*f
          OutH[((size_t)z * BCAP + row) * Hh + col] = f2bf(h);
        } else if constexpr (MODE == M_PROJ) {
          OutH[((size_t)z * BCAP + row) * Cc + col] = f2bf(v);
        } else {  // COMBINE -> y[b][t][c]
          OutF[((size_t)z * Tt + row) * Cc + col] = v;
        }
      }
}

// ---------------------------------------------------------------------------
// f32 -> bf16 conversion, 8 elements/thread/iter (b128 in, b128 out)
// ---------------------------------------------------------------------------
__global__ __launch_bounds__(256)
void cvt_f32_bf16(const float* __restrict__ in, __bf16* __restrict__ out, size_t n8) {
  size_t i = (size_t)blockIdx.x * blockDim.x + threadIdx.x;
  const size_t stride = (size_t)gridDim.x * blockDim.x;
  for (; i < n8; i += stride) {
    const float4 a = *(const float4*)(in + i * 8);
    const float4 b = *(const float4*)(in + i * 8 + 4);
    bf16x8 o;
    o[0] = f2bf(a.x); o[1] = f2bf(a.y); o[2] = f2bf(a.z); o[3] = f2bf(a.w);
    o[4] = f2bf(b.x); o[5] = f2bf(b.y); o[6] = f2bf(b.z); o[7] = f2bf(b.w);
    *(bf16x8*)(out + i * 8) = o;
  }
}

// ---------------------------------------------------------------------------
// dispatch softmax: over tokens T per (b, e*cap) column of g[B][T][ECAP].
// ---------------------------------------------------------------------------
__global__ __launch_bounds__(256)
void dispatch_softmax(const float* __restrict__ g, __bf16* __restrict__ out) {
  const int colIdx = blockIdx.x * 256 + threadIdx.x;
  const int b = blockIdx.y;
  const float* p = g   + (size_t)b * Tt * ECAP + colIdx;
  __bf16*      o = out + (size_t)b * Tt * ECAP + colIdx;
  float mx = -3.4e38f, sum = 0.f;
  for (int t = 0; t < Tt; ++t) {                      // online max/sum
    const float v = p[(size_t)t * ECAP];
    if (v > mx) { sum = sum * __expf(mx - v) + 1.f; mx = v; }
    else        { sum += __expf(v - mx); }
  }
  const float inv = 1.f / sum;
  for (int t = 0; t < Tt; ++t)
    o[(size_t)t * ECAP] = f2bf(__expf(p[(size_t)t * ECAP] - mx) * inv);
}

// ---------------------------------------------------------------------------
// collect softmax: over ECAP per (b,t) row; one block per row, vector loads.
// ---------------------------------------------------------------------------
__global__ __launch_bounds__(256)
void collect_softmax(const float* __restrict__ g, __bf16* __restrict__ out) {
  const size_t row = blockIdx.x;
  const float* p = g   + row * ECAP;
  __bf16*      o = out + row * ECAP;
  const int tid  = threadIdx.x;
  const int base = tid * 8;
  __shared__ float red[256];
  const float4 a = *(const float4*)(p + base);
  const float4 b = *(const float4*)(p + base + 4);
  float vals[8] = {a.x, a.y, a.z, a.w, b.x, b.y, b.z, b.w};
  float mx = -3.4e38f;
#pragma unroll
  for (int i = 0; i < 8; ++i) mx = fmaxf(mx, vals[i]);
  red[tid] = mx; __syncthreads();
  for (int s = 128; s > 0; s >>= 1) {
    if (tid < s) red[tid] = fmaxf(red[tid], red[tid + s]);
    __syncthreads();
  }
  mx = red[0]; __syncthreads();
  float sum = 0.f;
#pragma unroll
  for (int i = 0; i < 8; ++i) { vals[i] = __expf(vals[i] - mx); sum += vals[i]; }
  red[tid] = sum; __syncthreads();
  for (int s = 128; s > 0; s >>= 1) {
    if (tid < s) red[tid] += red[tid + s];
    __syncthreads();
  }
  const float inv = 1.f / red[0];
  bf16x8 ov;
#pragma unroll
  for (int i = 0; i < 8; ++i) ov[i] = f2bf(vals[i] * inv);
  *(bf16x8*)(o + base) = ov;
}

// ---------------------------------------------------------------------------
extern "C" void kernel_launch(void* const* d_in, const int* in_sizes, int n_in,
                              void* d_out, int out_size, void* d_ws, size_t ws_size,
                              hipStream_t stream) {
  const float* x   = (const float*)d_in[0];  // [B,T,C]
  const float* wr  = (const float*)d_in[1];  // [E,CAP,C]
  const float* wfc = (const float*)d_in[2];  // [E,C,H]
  const float* wg  = (const float*)d_in[3];  // [E,C,H]
  const float* wp  = (const float*)d_in[4];  // [E,H,C]
  float* y = (float*)d_out;                  // [B,T,C]

  char* ws = (char*)d_ws;
  size_t off = 0;
  auto alloc = [&](size_t bytes) -> void* {
    void* p = ws + off;
    off += (bytes + 255) & ~(size_t)255;
    return p;
  };
  const size_t nX  = (size_t)Bb * Tt * Cc;
  const size_t nWr = (size_t)Ee * CAP * Cc;
  const size_t nW  = (size_t)Ee * Cc * Hh;
  const size_t nG  = (size_t)Bb * Tt * ECAP;

  __bf16* xb   = (__bf16*)alloc(nX * 2);
  __bf16* wrb  = (__bf16*)alloc(nWr * 2);
  __bf16* wfcb = (__bf16*)alloc(nW * 2);
  __bf16* wgb  = (__bf16*)alloc(nW * 2);
  __bf16* wpb  = (__bf16*)alloc(nW * 2);
  float*  gbuf = (float*) alloc(nG * 4);
  __bf16* dsp  = (__bf16*)alloc(nG * 2);
  __bf16* col  = (__bf16*)alloc(nG * 2);
  __bf16* ei   = (__bf16*)alloc((size_t)Ee * BCAP * Cc * 2);
  __bf16* hb   = (__bf16*)alloc((size_t)Ee * BCAP * Hh * 2);
  __bf16* eo   = (__bf16*)alloc((size_t)Ee * BCAP * Cc * 2);

  // 1) bf16 conversions
  cvt_f32_bf16<<<2048, 256, 0, stream>>>(x,   xb,   nX  / 8);
  cvt_f32_bf16<<<512,  256, 0, stream>>>(wr,  wrb,  nWr / 8);
  cvt_f32_bf16<<<2048, 256, 0, stream>>>(wfc, wfcb, nW  / 8);
  cvt_f32_bf16<<<2048, 256, 0, stream>>>(wg,  wgb,  nW  / 8);
  cvt_f32_bf16<<<2048, 256, 0, stream>>>(wp,  wpb,  nW  / 8);

  // 2) router logits: g[b] = x[b] (T x C) @ Wr^T (C x ECAP)
  gemm_bf16<M_ROUTER><<<dim3(ECAP / BN, Tt / BM, Bb), 256, 0, stream>>>(
      xb, wrb, nullptr, gbuf, nullptr, Cc);

  // 3) softmaxes
  dispatch_softmax<<<dim3(ECAP / 256, Bb), 256, 0, stream>>>(gbuf, dsp);
  collect_softmax<<<dim3(Bb * Tt), 256, 0, stream>>>(gbuf, col);

  // 4) dispatch: ei[e][b*CAP+s][c] = sum_t disp[b][t][e*CAP+s] * x[b][t][c]
  gemm_bf16<M_DISPATCH><<<dim3(Cc / BN, ECAP / BM, Bb), 256, 0, stream>>>(
      dsp, xb, nullptr, nullptr, ei, Tt);

  // 5) fused GLU experts: h = silu(ei @ w_gate) * (ei @ w_fc)
  gemm_bf16<M_GLU><<<dim3(Hh / BN, BCAP / BM, Ee), 256, 0, stream>>>(
      ei, wgb, wfcb, nullptr, hb, Cc);

  // 6) projection: eo = h @ w_proj
  gemm_bf16<M_PROJ><<<dim3(Cc / BN, BCAP / BM, Ee), 256, 0, stream>>>(
      hb, wpb, nullptr, nullptr, eo, Hh);

  // 7) combine: y[b] = collect[b] (T x ECAP) @ eo_b (ECAP x C)
  gemm_bf16<M_COMBINE><<<dim3(Cc / BN, Tt / BM, Bb), 256, 0, stream>>>(
      col, eo, nullptr, y, nullptr, ECAP);
}
